// SFC_CAE_42425686949899
// MI455X (gfx1250) — matrix operand order repaired and checked by reading, not compiled, not run on hardware
//
#include <hip/hip_runtime.h>
#include <hip/hip_bf16.h>
#include <math.h>

// ---------------------------------------------------------------------------
// SFC-CAE forward for MI455X (gfx1250, wave32, WMMA).
// Convs / transposed convs (phase-decomposed) / FC stack all run on
// v_wmma_f32_16x16x32_f16. Fragments are staged in LDS as f16 with k = 16*(lane/16)+s
// (contiguous per lane half) so A/B packing is pure vectorized LDS loads.
// Conv kernels are templated on <CIN,COUT> so all index math is shifts and
// the channel loops fully unroll (no runtime-division sequences).
// ---------------------------------------------------------------------------

typedef __attribute__((ext_vector_type(16))) _Float16 v16h;
typedef __attribute__((ext_vector_type(8)))  float    v8f;

union Frag {
    v16h      v;
    _Float16  h[16];
    float2    f2[4];
    float4    f4[2];
};
union CFrag { v8f v; float f[8]; };

#define BATCH 16
#define NPTS  262144

// ---------------------------------------------------------------------------
// SFC 3-tap gather + tanh
// ---------------------------------------------------------------------------
__global__ void k_sp(const float* __restrict__ x, const int* __restrict__ ord,
                     const float* __restrict__ spw, const float* __restrict__ spb,
                     float* __restrict__ out, int N)
{
    int j = blockIdx.x * blockDim.x + threadIdx.x;
    int b = blockIdx.y;
    if (j >= N) return;
    int om = ord[j > 0 ? j - 1 : 0];
    int oo = ord[j];
    int op = ord[j < N - 1 ? j + 1 : N - 1];
    const float* xb = x + (size_t)b * N;
    float s = spw[j * 3 + 0] * xb[om] + spw[j * 3 + 1] * xb[oo]
            + spw[j * 3 + 2] * xb[op] + spb[j];
    out[(size_t)b * N + j] = tanhf(s);
}

// ---------------------------------------------------------------------------
// Strided conv1d (K=32, stride=4, pad=16) + bias + tanh as WMMA GEMM.
// One wave -> 64 output positions (4 N-tiles) x COUT(<=16).
//   A (16 x CIN*32): f16 weights in LDS, rows m>=COUT zeroed;
//     fragment = two 16B contiguous runs -> float4 LDS loads.
//   B (32 x 16 per ci-chunk): f16 window in LDS; fragment = 16 consecutive
//     halves at 8B alignment -> four float2 LDS loads.
// ---------------------------------------------------------------------------
template <int CIN, int COUT>
__global__ __launch_bounds__(32)
void k_conv_wmma(const float* __restrict__ in, const float* __restrict__ W,
                 const float* __restrict__ bias, float* __restrict__ out,
                 int Lin, int Lout, int outBStride)
{
    constexpr int WROW = CIN * 32;
    __shared__ __align__(16) _Float16 wsh[16 * WROW];      // [m][ci*32+k]
    __shared__ __align__(16) _Float16 winh[CIN * 284];     // [ci][4*63+32]
    const int n0   = blockIdx.x * 64;
    const int b    = blockIdx.y;
    const int base = n0 * 4 - 16;
    const int lane = threadIdx.x;

    // stage weights (f32 -> f16), pad M to 16 rows with zeros
    for (int idx = lane; idx < 16 * WROW; idx += 32) {
        int m = idx / WROW, r = idx % WROW;
        wsh[idx] = (_Float16)((m < COUT) ? W[m * WROW + r] : 0.f);
    }
    // stage input window (f32 -> f16), zero-padded at edges
    for (int idx = lane; idx < CIN * 284; idx += 32) {
        int ci = idx / 284, off = idx % 284;
        int pos = base + off;
        float v = 0.f;
        if (pos >= 0 && pos < Lin) v = in[((size_t)b * CIN + ci) * Lin + pos];
        winh[ci * 284 + off] = (_Float16)v;
    }
    __syncthreads();

    const int lg = lane >> 4, lm = lane & 15;
    v8f acc0 = {}, acc1 = {}, acc2 = {}, acc3 = {};
    #pragma unroll
    for (int ci = 0; ci < CIN; ++ci) {
        Frag a;
        const _Float16* ap = &wsh[lm * WROW + ci * 32];
        a.f4[0] = *(const float4*)(ap + 16 * lg);       // K = 16lg .. 16lg+7
        a.f4[1] = *(const float4*)(ap + 16 * lg + 8);   // K = 16lg+8 .. +7
        const _Float16* bp = &winh[ci * 284 + 4 * lm + 16 * lg];
        #pragma unroll
        for (int tt = 0; tt < 4; ++tt) {
            Frag bf;
            const _Float16* bt = bp + 64 * tt;          // 8B-aligned
            bf.f2[0] = *(const float2*)(bt + 0);
            bf.f2[1] = *(const float2*)(bt + 4);
            bf.f2[2] = *(const float2*)(bt + 8);
            bf.f2[3] = *(const float2*)(bt + 12);
            v8f* acc = (tt == 0) ? &acc0 : (tt == 1) ? &acc1 : (tt == 2) ? &acc2 : &acc3;
            *acc = __builtin_amdgcn_wmma_f32_16x16x32_f16(false, a.v, false, bf.v,
                                                          (short)0, *acc, false, false);
        }
    }
    #pragma unroll
    for (int tt = 0; tt < 4; ++tt) {
        CFrag cc; cc.v = (tt == 0) ? acc0 : (tt == 1) ? acc1 : (tt == 2) ? acc2 : acc3;
        #pragma unroll
        for (int r = 0; r < 8; ++r) {
            int m = r + 8 * lg;
            int t = n0 + 16 * tt + lm;
            if (m < COUT && t < Lout)
                out[(size_t)b * outBStride + (size_t)m * Lout + t] = tanhf(cc.f[r] + bias[m]);
        }
    }
}

// ---------------------------------------------------------------------------
// Transposed conv1d (lhs_dilation=4, K=32) via phase decomposition:
// for t%4==phase only taps k=k0+4j are live -> 8-tap stride-1 conv.
// Weights pre-flipped into LDS as aw[m][ci*8+jj] = Wt[ci,m,31-k0-4jj] (f16)
// -> vectorized A-frag loads. f16 window, 4 N-tiles (64 outputs of one phase) per block.
// ---------------------------------------------------------------------------
template <int CIN, int COUT>
__global__ __launch_bounds__(32)
void k_convT_wmma(const float* __restrict__ in, const float* __restrict__ Wt,
                  const float* __restrict__ bias, float* __restrict__ out,
                  int Lin, int Lout)
{
    constexpr int AROW = CIN * 8;
    __shared__ __align__(16) _Float16 awh[16 * AROW];      // [m][ci*8+jj]
    __shared__ __align__(16) _Float16 winh[CIN * 72];      // [ci][63+8]
    const int tbase = blockIdx.x * 256;
    const int phase = blockIdx.y;
    const int b     = blockIdx.z;
    const int k0    = (15 - phase) & 3;
    const int s0    = (tbase + phase - 15 + k0) >> 2;      // exact /4
    const int lane  = threadIdx.x;

    for (int idx = lane; idx < 16 * AROW; idx += 32) {
        int m = idx / AROW, rr = idx % AROW;
        int ci = rr >> 3, jj = rr & 7;
        float v = (m < COUT) ? Wt[(ci * COUT + m) * 32 + (31 - k0 - 4 * jj)] : 0.f;
        awh[idx] = (_Float16)v;
    }
    for (int idx = lane; idx < CIN * 71; idx += 32) {
        int ci = idx / 71, off = idx % 71;
        int s = s0 + off;
        float v = 0.f;
        if (s >= 0 && s < Lin) v = in[((size_t)b * CIN + ci) * Lin + s];
        winh[ci * 72 + off] = (_Float16)v;
    }
    __syncthreads();

    const int lg = lane >> 4, lm = lane & 15;
    v8f acc0 = {}, acc1 = {}, acc2 = {}, acc3 = {};
    constexpr int NCH = AROW / 32;                          // CIN/4
    #pragma unroll
    for (int c = 0; c < NCH; ++c) {
        Frag a;
        const _Float16* ap = &awh[lm * AROW + c * 32];
        a.f4[0] = *(const float4*)(ap + 16 * lg);
        a.f4[1] = *(const float4*)(ap + 16 * lg + 8);
        // B frag: halves j=0..7 -> row ci=4c+2lg, j=8..15 -> row 4c+2lg+1
        const _Float16* br0 = &winh[(4 * c + 2 * lg) * 72];
        const _Float16* br1 = br0 + 72;
        #pragma unroll
        for (int tt = 0; tt < 4; ++tt) {
            int n = lm + 16 * tt;
            Frag bf;
            #pragma unroll
            for (int j = 0; j < 8; ++j) {
                bf.h[j]     = br0[n + j];
                bf.h[j + 8] = br1[n + j];
            }
            v8f* acc = (tt == 0) ? &acc0 : (tt == 1) ? &acc1 : (tt == 2) ? &acc2 : &acc3;
            *acc = __builtin_amdgcn_wmma_f32_16x16x32_f16(false, a.v, false, bf.v,
                                                          (short)0, *acc, false, false);
        }
    }
    #pragma unroll
    for (int tt = 0; tt < 4; ++tt) {
        CFrag cc; cc.v = (tt == 0) ? acc0 : (tt == 1) ? acc1 : (tt == 2) ? acc2 : acc3;
        #pragma unroll
        for (int r = 0; r < 8; ++r) {
            int m = r + 8 * lg;
            int t = tbase + phase + 4 * (lm + 16 * tt);
            if (m < COUT && t < Lout)
                out[((size_t)b * COUT + m) * Lout + t] = tanhf(cc.f[r] + bias[m]);
        }
    }
}

// ---------------------------------------------------------------------------
// FC: out[b,o] = tanh(bias[o] + sum_k in[b,k]*W[o,k]), M = batch = 16.
// 8 waves split K; float4 global loads of f32 weights/activations (streamed
// once from HBM, converted to f16 in-register); LDS cross-wave reduction.
// ---------------------------------------------------------------------------
__global__ __launch_bounds__(256)
void k_fc_wmma(const float* __restrict__ in, const float* __restrict__ W,
               const float* __restrict__ bias, float* __restrict__ out,
               int Kin, int Nout)
{
    __shared__ float red[8 * 256];
    const int tid  = threadIdx.x;
    const int wave = tid >> 5;
    const int lane = tid & 31;
    const int lg = lane >> 4, lm = lane & 15;
    const int otile = blockIdx.x * 16;
    const int nchunks = (Kin + 31) / 32;

    v8f acc = {};
    for (int c = wave; c < nchunks; c += 8) {
        const int kbase = c * 32;
        const float* wrow = W + (size_t)(otile + lm) * Kin;
        const float* arow = in + (size_t)lm * Kin;
        __builtin_prefetch(wrow + kbase + 8 * 32, 0, 1);   // global_prefetch_b8
        Frag a, bf;
        if (kbase + 32 <= Kin) {
            // A-frag: halves j=0..15 -> K=kbase+16lg+j
            float ta[8], tc[8], tb[16];
            *(float4*)&ta[0] = *(const float4*)(arow + kbase + 16 * lg);
            *(float4*)&ta[4] = *(const float4*)(arow + kbase + 16 * lg + 4);
            *(float4*)&tc[0] = *(const float4*)(arow + kbase + 16 * lg + 8);
            *(float4*)&tc[4] = *(const float4*)(arow + kbase + 16 * lg + 12);
            // B-frag: halves j=0..15 -> K=kbase+16lg+j
            const float4* wv = (const float4*)(wrow + kbase + 16 * lg);
            *(float4*)&tb[0]  = wv[0];
            *(float4*)&tb[4]  = wv[1];
            *(float4*)&tb[8]  = wv[2];
            *(float4*)&tb[12] = wv[3];
            #pragma unroll
            for (int j = 0; j < 8; ++j) {
                a.h[j]     = (_Float16)ta[j];
                a.h[j + 8] = (_Float16)tc[j];
            }
            #pragma unroll
            for (int j = 0; j < 16; ++j) bf.h[j] = (_Float16)tb[j];
        } else {                                            // partial chunk (Kin=16)
            #pragma unroll
            for (int j = 0; j < 16; ++j) {
                int Ka = kbase + j + 16 * lg;
                a.h[j] = (_Float16)((Ka < Kin) ? arow[Ka] : 0.f);
                int Kb = kbase + j + 16 * lg;
                bf.h[j] = (_Float16)((Kb < Kin) ? wrow[Kb] : 0.f);
            }
        }
        acc = __builtin_amdgcn_wmma_f32_16x16x32_f16(false, a.v, false, bf.v,
                                                     (short)0, acc, false, false);
    }
    CFrag cc; cc.v = acc;
    #pragma unroll
    for (int r = 0; r < 8; ++r) red[wave * 256 + lane * 8 + r] = cc.f[r];
    __syncthreads();
    if (wave == 0) {
        #pragma unroll
        for (int r = 0; r < 8; ++r) {
            float s = 0.f;
            #pragma unroll
            for (int w2 = 0; w2 < 8; ++w2) s += red[w2 * 256 + lane * 8 + r];
            int m = r + 8 * lg;                 // batch row
            int o = otile + lm;                 // output feature
            out[(size_t)m * Nout + o] = tanhf(s + bias[o]);
        }
    }
}

// latent deinterleave: a_i[b,c,l] = hdec[b, 2*(c*1025+l) + i]
__global__ void k_unpack(const float* __restrict__ hdec, float* __restrict__ out, int sfc)
{
    int idx = blockIdx.x * blockDim.x + threadIdx.x;
    if (idx >= BATCH * 16400) return;
    int b = idx / 16400, q = idx % 16400;
    out[idx] = hdec[(size_t)b * 32800 + 2 * q + sfc];
}

// final blend: z = tanh(z0*fw0 + z1*fw1 + fb)
__global__ void k_final(const float* __restrict__ z0, const float* __restrict__ z1,
                        const float* __restrict__ fw, const float* __restrict__ fb,
                        float* __restrict__ out, int N)
{
    int j = blockIdx.x * blockDim.x + threadIdx.x;
    int b = blockIdx.y;
    if (j >= N) return;
    size_t id = (size_t)b * N + j;
    out[id] = tanhf(z0[id] * fw[j * 2 + 0] + z1[id] * fw[j * 2 + 1] + fb[j]);
}

// ---------------------------------------------------------------------------
// Host-side launch
// ---------------------------------------------------------------------------
struct ParamPtrs {
    const float *x;
    const int   *enc_ord, *dec_ord;
    const float *enc_sp_w, *enc_sp_b, *dec_sp_w, *dec_sp_b;
    const float *encW[2][4], *encB[2][4];
    const float *decW[2][4], *decB[2][4];
    const float *efcW[3], *efcB[3], *dfcW[3], *dfcB[3];
    const float *final_w, *final_b;
};

extern "C" void kernel_launch(void* const* d_in, const int* in_sizes, int n_in,
                              void* d_out, int out_size, void* d_ws, size_t ws_size,
                              hipStream_t stream) {
    if (n_in < 53) return;
    ParamPtrs P;
    auto F = [&](int k) { return (const float*)d_in[k]; };
    auto I = [&](int k) { return (const int*)d_in[k]; };

    if (in_sizes[0] == BATCH * NPTS) {
        // dict-insertion-order flattening: x, params{...}, enc_ord, dec_ord
        int k = 0;
        P.x = F(k++);
        P.enc_sp_w = F(k++); P.enc_sp_b = F(k++);
        for (int i = 0; i < 2; ++i) for (int j = 0; j < 4; ++j) { P.encW[i][j] = F(k++); P.encB[i][j] = F(k++); }
        for (int j = 0; j < 3; ++j) { P.efcW[j] = F(k++); P.efcB[j] = F(k++); }
        for (int j = 0; j < 3; ++j) { P.dfcW[j] = F(k++); P.dfcB[j] = F(k++); }
        for (int i = 0; i < 2; ++i) for (int j = 0; j < 4; ++j) { P.decW[i][j] = F(k++); P.decB[i][j] = F(k++); }
        P.dec_sp_w = F(k++); P.dec_sp_b = F(k++);
        P.final_w = F(k++); P.final_b = F(k++);
        P.enc_ord = I(k++); P.dec_ord = I(k++);
    } else {
        // jax tree_leaves alphabetical flattening: dec_ord, enc_ord, params(sorted), x
        int k = 0;
        P.dec_ord = I(k++); P.enc_ord = I(k++);
        for (int i = 0; i < 2; ++i) for (int j = 0; j < 4; ++j) { P.decW[i][j] = F(k++); P.decB[i][j] = F(k++); }
        for (int j = 0; j < 3; ++j) { P.dfcW[j] = F(k++); P.dfcB[j] = F(k++); }
        P.dec_sp_b = F(k++); P.dec_sp_w = F(k++);
        for (int i = 0; i < 2; ++i) for (int j = 0; j < 4; ++j) { P.encW[i][j] = F(k++); P.encB[i][j] = F(k++); }
        for (int j = 0; j < 3; ++j) { P.efcW[j] = F(k++); P.efcB[j] = F(k++); }
        P.enc_sp_b = F(k++); P.enc_sp_w = F(k++);
        P.final_b = F(k++); P.final_w = F(k++);
        P.x = F(k++);
    }

    // workspace layout (floats)
    const size_t SZ_PING = 4210688;              // > 16*16*16385
    float* ws    = (float*)d_ws;
    float* pingA = ws;
    float* pingB = pingA + SZ_PING;
    float* Z0    = pingB + SZ_PING;
    float* Z1    = Z0 + (size_t)BATCH * NPTS;
    float* h     = Z1 + (size_t)BATCH * NPTS;    // 16 x 32800
    float* hdec  = h + (size_t)BATCH * 32800;    // 16 x 32800
    float* t1    = hdec + (size_t)BATCH * 32800; // 16x1024
    float* t2    = t1 + 16 * 1024;               // 16x128
    float* t3    = t2 + 16 * 128;                // 16x16
    float* t4    = t3 + 16 * 16;                 // 16x128
    float* t5    = t4 + 16 * 128;                // 16x1024

    dim3 spGrid((NPTS + 255) / 256, BATCH);
    auto cg = [](int Lout) { return dim3((Lout + 63) / 64, BATCH); };
    auto tg = [](int Lout) { return dim3((Lout + 255) / 256, 4, BATCH); };

    // ---- encoder: sp -> conv chain (1->4->16->16->16) ----
    for (int i = 0; i < 2; ++i) {
        k_sp<<<spGrid, 256, 0, stream>>>(P.x, P.enc_ord + (size_t)i * NPTS,
                                         P.enc_sp_w + (size_t)i * NPTS * 3,
                                         P.enc_sp_b + (size_t)i * NPTS, pingA, NPTS);
        k_conv_wmma<1, 4><<<cg(65537), 32, 0, stream>>>(
            pingA, P.encW[i][0], P.encB[i][0], pingB, 262144, 65537, 4 * 65537);
        k_conv_wmma<4, 16><<<cg(16385), 32, 0, stream>>>(
            pingB, P.encW[i][1], P.encB[i][1], pingA, 65537, 16385, 16 * 16385);
        k_conv_wmma<16, 16><<<cg(4097), 32, 0, stream>>>(
            pingA, P.encW[i][2], P.encB[i][2], pingB, 16385, 4097, 16 * 4097);
        k_conv_wmma<16, 16><<<cg(1025), 32, 0, stream>>>(
            pingB, P.encW[i][3], P.encB[i][3], h + (size_t)i * 16400, 4097, 1025, 32800);
    }

    // ---- FC stack: 32800 -> 1024 -> 128 -> 16 -> 128 -> 1024 -> 32800 ----
    k_fc_wmma<<<1024 / 16, 256, 0, stream>>>(h,  P.efcW[0], P.efcB[0], t1, 32800, 1024);
    k_fc_wmma<<<128 / 16,  256, 0, stream>>>(t1, P.efcW[1], P.efcB[1], t2, 1024, 128);
    k_fc_wmma<<<16 / 16,   256, 0, stream>>>(t2, P.efcW[2], P.efcB[2], t3, 128, 16);
    k_fc_wmma<<<128 / 16,  256, 0, stream>>>(t3, P.dfcW[0], P.dfcB[0], t4, 16, 128);
    k_fc_wmma<<<1024 / 16, 256, 0, stream>>>(t4, P.dfcW[1], P.dfcB[1], t5, 128, 1024);
    k_fc_wmma<<<32800 / 16, 256, 0, stream>>>(t5, P.dfcW[2], P.dfcB[2], hdec, 1024, 32800);

    // ---- decoder: unpack -> convT chain (16->16->16->4->1) -> sp ----
    for (int i = 0; i < 2; ++i) {
        k_unpack<<<(BATCH * 16400 + 255) / 256, 256, 0, stream>>>(hdec, pingA, i);
        k_convT_wmma<16, 16><<<tg(4097), 32, 0, stream>>>(
            pingA, P.decW[i][0], P.decB[i][0], pingB, 1025, 4097);
        k_convT_wmma<16, 16><<<tg(16385), 32, 0, stream>>>(
            pingB, P.decW[i][1], P.decB[i][1], pingA, 4097, 16385);
        k_convT_wmma<16, 4><<<tg(65537), 32, 0, stream>>>(
            pingA, P.decW[i][2], P.decB[i][2], pingB, 16385, 65537);
        k_convT_wmma<4, 1><<<tg(262144), 32, 0, stream>>>(
            pingB, P.decW[i][3], P.decB[i][3], pingA, 65537, 262144);
        k_sp<<<spGrid, 256, 0, stream>>>(pingA, P.dec_ord + (size_t)i * NPTS,
                                         P.dec_sp_w + (size_t)i * NPTS * 3,
                                         P.dec_sp_b + (size_t)i * NPTS,
                                         (i == 0) ? Z0 : Z1, NPTS);
    }

    // ---- final blend ----
    k_final<<<spGrid, 256, 0, stream>>>(Z0, Z1, P.final_w, P.final_b,
                                        (float*)d_out, NPTS);
}
